// MyModel_63771674411489
// MI455X (gfx1250) — matrix operand order repaired and checked by reading, not compiled, not run on hardware
//
#include <hip/hip_runtime.h>

// ---------------------------------------------------------------------------
// GraphSAGE forward on MI455X (gfx1250, wave32, WMMA).
//   h0 = relu(x @ W_lin^T + b_lin)
//   h1 = (mean_agg(h0) @ Wl1^T + bl1) + h0 @ Wr1^T
//   h2 = (mean_agg(h1) @ Wl2^T + bl2) + h1 @ Wr2^T
//   out = relu(h2) @ W_cls^T
//
// v2 strategy: all WMMA operands are pre-converted to bf16 ONCE (element-wise
// kernels), so GEMM inner loops are pure packed-bf16 global loads + v_wmma —
// no per-tile f32->bf16 VALU storm. Scatter/aggregation stays fp32 atomics.
// ---------------------------------------------------------------------------

#define N_NODES     10000
#define N_EDGES     640000
#define IN_FEATS    256
#define H_FEATS     128
#define NUM_CLASSES 64

typedef __attribute__((ext_vector_type(16))) __bf16 v16bf;   // 32 B
typedef __attribute__((ext_vector_type(8)))  __bf16 v8bf;    // 16 B
typedef __attribute__((ext_vector_type(4)))  __bf16 v4bf;    // 8 B
typedef __attribute__((ext_vector_type(8)))  float  v8f;

__device__ __forceinline__ __bf16 f2bf(float f) {
  unsigned u = __builtin_bit_cast(unsigned, f);
  unsigned r = u + 0x7FFFu + ((u >> 16) & 1u);      // round-to-nearest-even
  unsigned short h = (unsigned short)(r >> 16);
  return __builtin_bit_cast(__bf16, h);
}

// ---------------------------------------------------------------------------
// Element-wise f32 -> bf16 (float4 -> 4x bf16 per thread).
// ---------------------------------------------------------------------------
__launch_bounds__(256)
__global__ void cvt_f32_to_bf16(const float* __restrict__ in,
                                __bf16* __restrict__ out, int n4) {
  int t = blockIdx.x * blockDim.x + threadIdx.x;
  if (t < n4) {
    float4 v = ((const float4*)in)[t];
    v4bf o = { f2bf(v.x), f2bf(v.y), f2bf(v.z), f2bf(v.w) };
    ((v4bf*)out)[t] = o;
  }
}

// mean_bf[node][c..c+3] = bf16(agg[node][c..c+3] / max(deg[node],1))
__launch_bounds__(256)
__global__ void mean_to_bf16(const float* __restrict__ agg,
                             const float* __restrict__ deg,
                             __bf16* __restrict__ out) {
  int t = blockIdx.x * blockDim.x + threadIdx.x;        // N_NODES * 32 threads
  if (t < N_NODES * (H_FEATS / 4)) {
    const int node = t >> 5;
    const float invd = 1.0f / fmaxf(deg[node], 1.0f);
    float4 v = ((const float4*)agg)[t];
    v4bf o = { f2bf(v.x * invd), f2bf(v.y * invd),
               f2bf(v.z * invd), f2bf(v.w * invd) };
    ((v4bf*)out)[t] = o;
  }
}

// ---------------------------------------------------------------------------
// Fragment loads: packed bf16 straight from memory (no conversion).
// A-fragment (16x32): lane&15 = row M; half = lane>>4 picks
// K chunks {0..15} / {16..31} (matches B). Two 16-B vector loads.
// ---------------------------------------------------------------------------
__device__ __forceinline__ v16bf load_fragA(const __bf16* __restrict__ row,
                                            int kbase) {
  v8bf lo = *(const v8bf*)(row + kbase);
  v8bf hi = *(const v8bf*)(row + kbase + 8);
  return __builtin_shufflevector(lo, hi, 0, 1, 2, 3, 4, 5, 6, 7,
                                          8, 9, 10, 11, 12, 13, 14, 15);
}

// B-fragment (32x16, contiguous K order, as in A): lane&15 = col N; lanes 0-15 hold
// K=0..15, lanes 16-31 hold K=16..31. One 32-B vector load (kbase pre-offset).
__device__ __forceinline__ v16bf load_fragB(const __bf16* __restrict__ wrow,
                                            int kbase) {
  return *(const v16bf*)(wrow + kbase);
}

// ---------------------------------------------------------------------------
// C = act(A[M,K] @ W[N,K]^T + bias); one 16x16 tile per wave (wave = N-tile).
// Optionally stores f32 (for scatter) and/or bf16 (next layer's A operand).
// ---------------------------------------------------------------------------
template<int K, int N, bool RELU_OUT, bool STORE_F32, bool STORE_BF16>
__launch_bounds__(256)
__global__ void gemm_bf16(const __bf16* __restrict__ A,
                          const __bf16* __restrict__ W,
                          const float* __restrict__ bias,
                          float* __restrict__ Cf,
                          __bf16* __restrict__ Cb) {
  const int lane  = threadIdx.x & 31;
  const int tileM = blockIdx.x * 16;
  const int tileN = (threadIdx.x >> 5) * 16;
  const int ln    = lane & 15;
  const int half  = lane >> 4;

  const __bf16* arow = A + (size_t)(tileM + ln) * K;
  const __bf16* wrow = W + (size_t)(tileN + ln) * K;

  v8f acc = {};
#pragma unroll
  for (int k0 = 0; k0 < K; k0 += 32) {
    v16bf a = load_fragA(arow, k0 + half * 16);
    v16bf b = load_fragB(wrow, k0 + half * 16);
    acc = __builtin_amdgcn_wmma_f32_16x16x32_bf16(false, a, false, b,
                                                  (short)0, acc, false, false);
  }

  const int col = tileN + ln;
  const int mb  = tileM + half * 8;
  const float bz = bias ? bias[col] : 0.0f;
#pragma unroll
  for (int r = 0; r < 8; ++r) {
    float v = acc[r] + bz;
    if (RELU_OUT) v = fmaxf(v, 0.0f);
    const size_t idx = (size_t)(mb + r) * N + col;
    if (STORE_F32)  Cf[idx] = v;
    if (STORE_BF16) Cb[idx] = f2bf(v);
  }
}

// ---------------------------------------------------------------------------
// out = mean_bf @ Wl^T + bl + h_bf @ Wr^T   (K = N = H_FEATS = 128)
// ---------------------------------------------------------------------------
template<bool RELU_OUT, bool STORE_F32>
__launch_bounds__(256)
__global__ void sage_combine(const __bf16* __restrict__ mean_bf,
                             const __bf16* __restrict__ h_bf,
                             const __bf16* __restrict__ Wl,
                             const float* __restrict__ bl,
                             const __bf16* __restrict__ Wr,
                             float* __restrict__ outF,
                             __bf16* __restrict__ outB) {
  const int lane  = threadIdx.x & 31;
  const int tileM = blockIdx.x * 16;
  const int tileN = (threadIdx.x >> 5) * 16;
  const int ln    = lane & 15;
  const int half  = lane >> 4;

  const __bf16* mrow  = mean_bf + (size_t)(tileM + ln) * H_FEATS;
  const __bf16* hrow  = h_bf    + (size_t)(tileM + ln) * H_FEATS;
  const __bf16* wlrow = Wl      + (size_t)(tileN + ln) * H_FEATS;
  const __bf16* wrrow = Wr      + (size_t)(tileN + ln) * H_FEATS;

  v8f acc = {};
#pragma unroll
  for (int k0 = 0; k0 < H_FEATS; k0 += 32) {
    v16bf a = load_fragA(mrow, k0 + half * 16);
    v16bf b = load_fragB(wlrow, k0 + half * 16);
    acc = __builtin_amdgcn_wmma_f32_16x16x32_bf16(false, a, false, b,
                                                  (short)0, acc, false, false);
  }
#pragma unroll
  for (int k0 = 0; k0 < H_FEATS; k0 += 32) {
    v16bf a = load_fragA(hrow, k0 + half * 16);
    v16bf b = load_fragB(wrrow, k0 + half * 16);
    acc = __builtin_amdgcn_wmma_f32_16x16x32_bf16(false, a, false, b,
                                                  (short)0, acc, false, false);
  }

  const int col = tileN + ln;
  const int mb  = tileM + half * 8;
  const float bz = bl[col];
#pragma unroll
  for (int r = 0; r < 8; ++r) {
    float v = acc[r] + bz;
    if (RELU_OUT) v = fmaxf(v, 0.0f);
    const size_t idx = (size_t)(mb + r) * H_FEATS + col;
    if (STORE_F32) outF[idx] = v;
    outB[idx] = f2bf(v);
  }
}

// ---------------------------------------------------------------------------
// Aggregation (fp32 for accuracy; L2-resident: h + agg ~10 MB << 192 MB L2).
// ---------------------------------------------------------------------------
__launch_bounds__(256)
__global__ void degree_f32(const long long* __restrict__ dst,
                           float* __restrict__ deg) {
  int t = blockIdx.x * blockDim.x + threadIdx.x;
  if (t < N_EDGES) atomicAdd(&deg[(int)dst[t]], 1.0f);
}

// agg[dst] += h[src]; one (edge, 4-float chunk) per thread, grid-stride.
__launch_bounds__(256)
__global__ void scatter_add_f32(const float* __restrict__ h,
                                const long long* __restrict__ src,
                                const long long* __restrict__ dst,
                                float* __restrict__ agg) {
  const long long total = (long long)N_EDGES * (H_FEATS / 4);
  for (long long t = (long long)blockIdx.x * blockDim.x + threadIdx.x;
       t < total; t += (long long)gridDim.x * blockDim.x) {
    const int e = (int)(t >> 5);           // 32 chunks of 4 floats per edge
    const int c = ((int)t & 31) * 4;
    const int s = (int)src[e];
    const int d = (int)dst[e];
    const float4 v = *(const float4*)(h + (size_t)s * H_FEATS + c);
    float* p = agg + (size_t)d * H_FEATS + c;
    atomicAdd(p + 0, v.x);
    atomicAdd(p + 1, v.y);
    atomicAdd(p + 2, v.z);
    atomicAdd(p + 3, v.w);
  }
}

// ---------------------------------------------------------------------------
extern "C" void kernel_launch(void* const* d_in, const int* in_sizes, int n_in,
                              void* d_out, int out_size, void* d_ws, size_t ws_size,
                              hipStream_t stream) {
  (void)in_sizes; (void)n_in; (void)out_size; (void)ws_size;
  const float*     x     = (const float*)d_in[0];
  const long long* ei    = (const long long*)d_in[1];   // int64 [2, E]
  const float*     W_lin = (const float*)d_in[2];
  const float*     b_lin = (const float*)d_in[3];
  const float*     Wl1   = (const float*)d_in[4];
  const float*     bl1   = (const float*)d_in[5];
  const float*     Wr1   = (const float*)d_in[6];
  const float*     Wl2   = (const float*)d_in[7];
  const float*     bl2   = (const float*)d_in[8];
  const float*     Wr2   = (const float*)d_in[9];
  const float*     Wcls  = (const float*)d_in[10];
  const long long* src = ei;
  const long long* dst = ei + N_EDGES;

  const size_t NH = (size_t)N_NODES * H_FEATS;          // 1.28M

  // Workspace carve-up (256-B aligned blocks).
  char* p = (char*)d_ws;
  auto alloc = [&](size_t bytes) -> void* {
    void* r = (void*)p;
    p += (bytes + 255) & ~(size_t)255;
    return r;
  };
  __bf16* x_bf    = (__bf16*)alloc((size_t)N_NODES * IN_FEATS * 2);   // 5.12 MB
  __bf16* wlin_bf = (__bf16*)alloc((size_t)H_FEATS * IN_FEATS * 2);
  __bf16* wl1_bf  = (__bf16*)alloc((size_t)H_FEATS * H_FEATS * 2);
  __bf16* wr1_bf  = (__bf16*)alloc((size_t)H_FEATS * H_FEATS * 2);
  __bf16* wl2_bf  = (__bf16*)alloc((size_t)H_FEATS * H_FEATS * 2);
  __bf16* wr2_bf  = (__bf16*)alloc((size_t)H_FEATS * H_FEATS * 2);
  __bf16* wcls_bf = (__bf16*)alloc((size_t)NUM_CLASSES * H_FEATS * 2);
  float*  h0_f    = (float*)alloc(NH * 4);                            // 5.12 MB
  __bf16* h0_bf   = (__bf16*)alloc(NH * 2);
  float*  h1_f    = (float*)alloc(NH * 4);
  __bf16* h1_bf   = (__bf16*)alloc(NH * 2);
  __bf16* h2_bf   = (__bf16*)alloc(NH * 2);                           // relu'd
  float*  agg     = (float*)alloc(NH * 4);
  __bf16* mean_bf = (__bf16*)alloc(NH * 2);
  float*  deg     = (float*)alloc((size_t)N_NODES * 4);

  const int MT = N_NODES / 16;    // 625 exact M-tiles

  // --- one-time conversions (element-wise, memory-bound) -------------------
  auto cvt = [&](const float* in, __bf16* out, int n) {
    int n4 = n / 4;
    cvt_f32_to_bf16<<<(n4 + 255) / 256, 256, 0, stream>>>(in, out, n4);
  };
  cvt(x,     x_bf,    N_NODES * IN_FEATS);
  cvt(W_lin, wlin_bf, H_FEATS * IN_FEATS);
  cvt(Wl1,   wl1_bf,  H_FEATS * H_FEATS);
  cvt(Wr1,   wr1_bf,  H_FEATS * H_FEATS);
  cvt(Wl2,   wl2_bf,  H_FEATS * H_FEATS);
  cvt(Wr2,   wr2_bf,  H_FEATS * H_FEATS);
  cvt(Wcls,  wcls_bf, NUM_CLASSES * H_FEATS);

  // --- degree (computed once, reused for both layers) ----------------------
  hipMemsetAsync(deg, 0, N_NODES * sizeof(float), stream);
  degree_f32<<<(N_EDGES + 255) / 256, 256, 0, stream>>>(dst, deg);

  // --- h0 = relu(x @ W_lin^T + b_lin) --------------------------------------
  gemm_bf16<IN_FEATS, H_FEATS, true, true, true>
      <<<MT, 256, 0, stream>>>(x_bf, wlin_bf, b_lin, h0_f, h0_bf);

  // --- layer 1 -------------------------------------------------------------
  hipMemsetAsync(agg, 0, NH * sizeof(float), stream);
  scatter_add_f32<<<2048, 256, 0, stream>>>(h0_f, src, dst, agg);
  mean_to_bf16<<<(N_NODES * 32 + 255) / 256, 256, 0, stream>>>(agg, deg, mean_bf);
  sage_combine<false, true>
      <<<MT, 256, 0, stream>>>(mean_bf, h0_bf, wl1_bf, bl1, wr1_bf, h1_f, h1_bf);

  // --- layer 2 (output only feeds classifier via relu -> bf16-only store) --
  hipMemsetAsync(agg, 0, NH * sizeof(float), stream);
  scatter_add_f32<<<2048, 256, 0, stream>>>(h1_f, src, dst, agg);
  mean_to_bf16<<<(N_NODES * 32 + 255) / 256, 256, 0, stream>>>(agg, deg, mean_bf);
  sage_combine<true, false>
      <<<MT, 256, 0, stream>>>(mean_bf, h1_bf, wl2_bf, bl2, wr2_bf, nullptr, h2_bf);

  // --- out = relu(h2) @ W_cls^T  (N=64 -> 4 waves per block) ---------------
  gemm_bf16<H_FEATS, NUM_CLASSES, false, true, false>
      <<<MT, 128, 0, stream>>>(h2_bf, wcls_bf, nullptr, (float*)d_out, nullptr);
}
